// Model_3478923510497
// MI455X (gfx1250) — matrix operand order repaired and checked by reading, not compiled, not run on hardware
//
#include <hip/hip_runtime.h>

#define NN      50000
#define EE      800000
#define DIN     128
#define HID     64
#define HEADS   4
#define DOUT    128
#define LAYERS  8
#define NUM_SEL 1024
#define NEG_SLOPE 0.2f

typedef __attribute__((ext_vector_type(16))) _Float16 v16h;
typedef __attribute__((ext_vector_type(8)))  _Float16 v8h;
typedef __attribute__((ext_vector_type(8)))  float    v8f;

// ---------------------------------------------------------------------------
// Weight pre-pass: W[K x Nc] f32 row-major  ->  Wt[Nc x K] f16 (transposed)
// ---------------------------------------------------------------------------
__global__ void transpose_to_f16(const float* __restrict__ W,
                                 _Float16* __restrict__ Wt,
                                 int K, int Nc)
{
    int tid = blockIdx.x * blockDim.x + threadIdx.x;
    if (tid >= K * Nc) return;
    int n = tid / K;
    int k = tid % K;
    Wt[(size_t)n * K + k] = (_Float16)W[(size_t)k * Nc + n];
}

// ---------------------------------------------------------------------------
// WMMA GEMM:  Y[M x Nc] = concat(A0 | A1)[M x K] * W[K x Nc] + bias
//  - Wt is the f16-transposed weight [Nc x K]
//  - whole Wt staged into LDS with global_load_async_to_lds_b128 (ASYNCcnt)
//  - one wave32 computes a 16(M) x 64(N) strip: 1 A fragment -> 4 WMMAs/k-step
//  - LDS row stride = K+8 halves => 272B rows, conflict-free ds_load_b128
// ---------------------------------------------------------------------------
__global__ void gemm_wmma_f16(const float* __restrict__ A0,
                              const float* __restrict__ A1,
                              int lda,
                              const _Float16* __restrict__ Wt, // [Nc x K] f16
                              const float* __restrict__ bias,  // [Nc]
                              float* __restrict__ Y,
                              int M, int K, int Nc)
{
    __shared__ __align__(16) _Float16 wt[256 * 136]; // max Nc=256, stride<=136

    const int stride = K + 8;

    // ---- cooperative async stage of Wt into LDS (16B chunks) ----
    const int cpr    = K >> 3;           // chunks per row
    const int chunks = Nc * cpr;
    for (int id = threadIdx.x; id < chunks; id += blockDim.x) {
        const int row = id / cpr;
        const int c8  = id % cpr;
        unsigned      ldsoff = (unsigned)(uintptr_t)(&wt[row * stride + c8 * 8]);
        unsigned long gaddr  = (unsigned long)(uintptr_t)(Wt + (size_t)id * 8);
        asm volatile("global_load_async_to_lds_b128 %0, %1, off"
                     :: "v"(ldsoff), "v"(gaddr) : "memory");
    }
    asm volatile("s_wait_asynccnt 0x0" ::: "memory");
    __syncthreads();

    const int lane = threadIdx.x & 31;
    const int wave = threadIdx.x >> 5;
    const int unit = blockIdx.x * (blockDim.x >> 5) + wave;
    const int nstrips = Nc >> 6;                 // 64-wide N strips
    const int nunits  = (M >> 4) * nstrips;
    if (unit < nunits) {
        const int m0 = (unit / nstrips) << 4;
        const int n0 = (unit % nstrips) << 6;

        const int mrow = lane & 15;
        const int hi   = lane >> 4;

        v8f acc[4] = {};
        for (int k0 = 0; k0 < K; k0 += 32) {
            const float* Abase = (k0 < 64) ? A0 : A1;
            const int kk = k0 & 63;

            // A fragment (ISA 16-bit A layout): K groups {hi*8..+7},{16+hi*8..+7}
            const float* arow = Abase + (size_t)(m0 + mrow) * lda + kk;
            const float4 q0 = ((const float4*)(arow + hi * 8))[0];
            const float4 q1 = ((const float4*)(arow + hi * 8))[1];
            const float4 q2 = ((const float4*)(arow + 16 + hi * 8))[0];
            const float4 q3 = ((const float4*)(arow + 16 + hi * 8))[1];
            v16h a;
            a[0]=(_Float16)q0.x;  a[1]=(_Float16)q0.y;  a[2]=(_Float16)q0.z;  a[3]=(_Float16)q0.w;
            a[4]=(_Float16)q1.x;  a[5]=(_Float16)q1.y;  a[6]=(_Float16)q1.z;  a[7]=(_Float16)q1.w;
            a[8]=(_Float16)q2.x;  a[9]=(_Float16)q2.y;  a[10]=(_Float16)q2.z; a[11]=(_Float16)q2.w;
            a[12]=(_Float16)q3.x; a[13]=(_Float16)q3.y; a[14]=(_Float16)q3.z; a[15]=(_Float16)q3.w;

            // 4 B fragments from LDS; K groups {hi*8..+7},{16+hi*8..+7} (matches A)
            #pragma unroll
            for (int t = 0; t < 4; ++t) {
                const _Float16* bp = &wt[(n0 + t * 16 + mrow) * stride + k0];
                v8h blo = *(const v8h*)(bp + hi * 8);
                v8h bhi = *(const v8h*)(bp + 16 + hi * 8);
                v16h b = __builtin_shufflevector(blo, bhi,
                        0,1,2,3,4,5,6,7,8,9,10,11,12,13,14,15);
                acc[t] = __builtin_amdgcn_wmma_f32_16x16x32_f16(
                    false, a, false, b, (short)0, acc[t], false, false);
            }
        }

        // store: VGPR r -> row m0 + hi*8 + r, col n0 + t*16 + mrow
        #pragma unroll
        for (int t = 0; t < 4; ++t) {
            const float bv = bias[n0 + t * 16 + mrow];
            float* yout = Y + (size_t)(m0 + hi * 8) * Nc + n0 + t * 16 + mrow;
            #pragma unroll
            for (int r = 0; r < 8; ++r)
                yout[(size_t)r * Nc] = acc[t][r] + bv;
        }
    }
}

// ---------------------------------------------------------------------------
// Order-preserving float <-> uint transforms for atomic max
// ---------------------------------------------------------------------------
__device__ __forceinline__ unsigned ford(float f) {
    unsigned u = __float_as_uint(f);
    return (u & 0x80000000u) ? ~u : (u | 0x80000000u);
}
__device__ __forceinline__ float ford_inv(unsigned u) {
    u = (u & 0x80000000u) ? (u ^ 0x80000000u) : ~u;
    return __uint_as_float(u);
}

__global__ void zero_u32(unsigned* __restrict__ p, size_t n) {
    size_t i = (size_t)blockIdx.x * blockDim.x + threadIdx.x;
    size_t stride = (size_t)gridDim.x * blockDim.x;
    for (; i < n; i += stride) p[i] = 0u;
}

// e[eid,h] = att[h,:] . leaky_relu(xl[src,h,:] + xr[dst,h,:]); atomic segment max
__global__ void edge_logits(const float* __restrict__ xl,
                            const float* __restrict__ xr,
                            const int* __restrict__ src,
                            const int* __restrict__ dst,
                            const float* __restrict__ att,
                            float* __restrict__ e,
                            unsigned* __restrict__ mmax)
{
    const long long tid = (long long)blockIdx.x * blockDim.x + threadIdx.x;
    if (tid >= (long long)EE * HEADS) return;
    const int eid = (int)(tid >> 2);
    const int h   = (int)(tid & 3);
    const int s = src[eid], d = dst[eid];
    const float4* pl = (const float4*)(xl + (size_t)s * (HEADS * HID) + h * HID);
    const float4* pr = (const float4*)(xr + (size_t)d * (HEADS * HID) + h * HID);
    const float4* pa = (const float4*)(att + h * HID);
    float acc = 0.f;
    #pragma unroll
    for (int c = 0; c < HID / 4; ++c) {
        float4 a = pl[c], b = pr[c], w = pa[c];
        float v;
        v = a.x + b.x; v = (v > 0.f) ? v : NEG_SLOPE * v; acc += w.x * v;
        v = a.y + b.y; v = (v > 0.f) ? v : NEG_SLOPE * v; acc += w.y * v;
        v = a.z + b.z; v = (v > 0.f) ? v : NEG_SLOPE * v; acc += w.z * v;
        v = a.w + b.w; v = (v > 0.f) ? v : NEG_SLOPE * v; acc += w.w * v;
    }
    e[(size_t)eid * HEADS + h] = acc;
    atomicMax(&mmax[(size_t)d * HEADS + h], ford(acc));
}

// ex = exp(e - m[dst]); denom[dst] += ex  (ex overwrites e)
__global__ void edge_exp(float* __restrict__ e,
                         const int* __restrict__ dst,
                         const unsigned* __restrict__ mmax,
                         float* __restrict__ denom)
{
    const long long tid = (long long)blockIdx.x * blockDim.x + threadIdx.x;
    if (tid >= (long long)EE * HEADS) return;
    const int eid = (int)(tid >> 2);
    const int h   = (int)(tid & 3);
    const int d = dst[eid];
    const float m = ford_inv(mmax[(size_t)d * HEADS + h]);
    const float ex = expf(e[(size_t)eid * HEADS + h] - m);
    e[(size_t)eid * HEADS + h] = ex;
    atomicAdd(&denom[(size_t)d * HEADS + h], ex);
}

// out[dst,h,:] += (ex / denom[dst,h]) * xl[src,h,:]
__global__ void edge_scatter(const float* __restrict__ e,
                             const float* __restrict__ xl,
                             const int* __restrict__ src,
                             const int* __restrict__ dst,
                             const float* __restrict__ denom,
                             float* __restrict__ out)
{
    const long long tid = (long long)blockIdx.x * blockDim.x + threadIdx.x;
    if (tid >= (long long)EE * HEADS) return;
    const int eid = (int)(tid >> 2);
    const int h   = (int)(tid & 3);
    const int s = src[eid], d = dst[eid];
    const float alpha = e[(size_t)eid * HEADS + h] /
                        (denom[(size_t)d * HEADS + h] + 1e-16f);
    const float4* pl = (const float4*)(xl + (size_t)s * (HEADS * HID) + h * HID);
    float* po = out + (size_t)d * (HEADS * HID) + h * HID;
    #pragma unroll
    for (int c = 0; c < HID / 4; ++c) {
        float4 v = pl[c];
        atomicAdd(po + 4 * c + 0, alpha * v.x);
        atomicAdd(po + 4 * c + 1, alpha * v.y);
        atomicAdd(po + 4 * c + 2, alpha * v.z);
        atomicAdd(po + 4 * c + 3, alpha * v.w);
    }
}

// h[n,c] = tanh( sum_h out[n,h,c] + sum_h conv_bias[h,c] )
__global__ void node_update(const float* __restrict__ out,
                            const float* __restrict__ cb,
                            float* __restrict__ h)
{
    const long long tid = (long long)blockIdx.x * blockDim.x + threadIdx.x;
    if (tid >= (long long)NN * HID) return;
    const int n = (int)(tid >> 6);
    const int c = (int)(tid & 63);
    const float* po = out + (size_t)n * (HEADS * HID);
    float sum = po[c] + po[HID + c] + po[2 * HID + c] + po[3 * HID + c]
              + cb[c] + cb[HID + c] + cb[2 * HID + c] + cb[3 * HID + c];
    h[(size_t)n * HID + c] = tanhf(sum);
}

// Stable masked select (ascending node order) into hg[NUM_SEL x HID]
__global__ void select_rows(const int* __restrict__ state,
                            const float* __restrict__ h,
                            float* __restrict__ hg)
{
    __shared__ int sdata[1024];
    __shared__ int base;
    if (threadIdx.x == 0) base = 0;
    __syncthreads();

    for (int start = 0; start < NN; start += 1024) {
        const int i = start + (int)threadIdx.x;
        const int v = (i < NN) ? state[i] : 0;
        sdata[threadIdx.x] = v;
        __syncthreads();
        for (int off = 1; off < 1024; off <<= 1) {
            int t = (threadIdx.x >= (unsigned)off) ? sdata[threadIdx.x - off] : 0;
            __syncthreads();
            sdata[threadIdx.x] += t;
            __syncthreads();
        }
        const int b0 = base;
        const int incl = sdata[threadIdx.x];
        if (v) {
            const int rank = b0 + incl - 1;
            if (rank < NUM_SEL) {
                const float4* s4 = (const float4*)(h + (size_t)i * HID);
                float4* d4 = (float4*)(hg + (size_t)rank * HID);
                #pragma unroll
                for (int c = 0; c < HID / 4; ++c) d4[c] = s4[c];
            }
        }
        __syncthreads();
        if (threadIdx.x == 0) base = b0 + sdata[1023];
        __syncthreads();
    }
}

// ---------------------------------------------------------------------------
extern "C" void kernel_launch(void* const* d_in, const int* in_sizes, int n_in,
                              void* d_out, int out_size, void* d_ws, size_t ws_size,
                              hipStream_t stream)
{
    const float* feat  = (const float*)d_in[0];
    const int*   edge  = (const int*)d_in[1];   // [2, E]
    const int*   state = (const int*)d_in[2];
    const float* W_in  = (const float*)d_in[3];
    const float* b_in  = (const float*)d_in[4];
    const float* Wl    = (const float*)d_in[5];
    const float* bl    = (const float*)d_in[6];
    const float* Wr    = (const float*)d_in[7];
    const float* br    = (const float*)d_in[8];
    const float* att   = (const float*)d_in[9];
    const float* cb    = (const float*)d_in[10];
    const float* Wg    = (const float*)d_in[11];
    const float* bg    = (const float*)d_in[12];
    const int* src = edge;
    const int* dst = edge + EE;

    // workspace carve-up (mmax|denom|out contiguous -> one zero pass)
    float* p   = (float*)d_ws;
    float* h0  = p;  p += (size_t)NN * HID;
    float* h   = p;  p += (size_t)NN * HID;
    float* xl  = p;  p += (size_t)NN * HEADS * HID;
    float* xr  = p;  p += (size_t)NN * HEADS * HID;
    float* e   = p;  p += (size_t)EE * HEADS;
    unsigned* mmax = (unsigned*)p; p += (size_t)NN * HEADS;
    float* denom = p; p += (size_t)NN * HEADS;
    float* outb  = p; p += (size_t)NN * (HEADS * HID);
    float* hg    = p; p += (size_t)NUM_SEL * HID;
    // f16 transposed weights
    _Float16* hp   = (_Float16*)p;
    _Float16* wtIn = hp; hp += (size_t)HID * DIN;                  // [64 x 128]
    _Float16* wtL  = hp; hp += (size_t)(HEADS * HID) * (2 * HID);  // [256 x 128]
    _Float16* wtR  = hp; hp += (size_t)(HEADS * HID) * (2 * HID);  // [256 x 128]
    _Float16* wtG  = hp; hp += (size_t)DOUT * HID;                 // [128 x 64]
    const size_t needBytes = (size_t)((char*)hp - (char*)d_ws);
    if (ws_size < needBytes) return;

    auto cdiv = [](long long a, long long b) { return (int)((a + b - 1) / b); };

    // pre-convert weights: f32 [K x Nc] -> f16 transposed [Nc x K]
    transpose_to_f16<<<cdiv(DIN * HID, 256), 256, 0, stream>>>(W_in, wtIn, DIN, HID);
    transpose_to_f16<<<cdiv(2 * HID * HEADS * HID, 256), 256, 0, stream>>>(Wl, wtL, 2 * HID, HEADS * HID);
    transpose_to_f16<<<cdiv(2 * HID * HEADS * HID, 256), 256, 0, stream>>>(Wr, wtR, 2 * HID, HEADS * HID);
    transpose_to_f16<<<cdiv(HID * DOUT, 256), 256, 0, stream>>>(Wg, wtG, HID, DOUT);

    // h0 = feat @ W_in + b_in   (M=50000, K=128, Nc=64 -> 1 strip of 64)
    {
        int units = (NN / 16) * (HID / 64);
        gemm_wmma_f16<<<cdiv(units, 8), 256, 0, stream>>>(
            feat, feat + 64, DIN, wtIn, b_in, h0, NN, DIN, HID);
    }

    const int eThreads = 256;
    const int eGrid = cdiv((long long)EE * HEADS, eThreads);
    const size_t zWords = (size_t)NN * HEADS + (size_t)NN * HEADS
                        + (size_t)NN * (HEADS * HID);

    for (int l = 0; l < LAYERS; ++l) {
        const float* xin = (l == 0) ? h0 : h;

        zero_u32<<<2048, 256, 0, stream>>>(mmax, zWords);

        // xl/xr = [xin|h0] @ W{l,r} + b{l,r}   (K=128, Nc=256 -> 4 strips)
        int units = (NN / 16) * ((HEADS * HID) / 64);
        gemm_wmma_f16<<<cdiv(units, 8), 256, 0, stream>>>(
            xin, h0, HID, wtL, bl, xl, NN, 2 * HID, HEADS * HID);
        gemm_wmma_f16<<<cdiv(units, 8), 256, 0, stream>>>(
            xin, h0, HID, wtR, br, xr, NN, 2 * HID, HEADS * HID);

        edge_logits<<<eGrid, eThreads, 0, stream>>>(xl, xr, src, dst, att, e, mmax);
        edge_exp<<<eGrid, eThreads, 0, stream>>>(e, dst, mmax, denom);
        edge_scatter<<<eGrid, eThreads, 0, stream>>>(e, xl, src, dst, denom, outb);
        node_update<<<cdiv((long long)NN * HID, 256), 256, 0, stream>>>(outb, cb, h);
    }

    select_rows<<<1, 1024, 0, stream>>>(state, h, hg);

    // d_out = hg @ Wg + bg   (M=1024, K=64, Nc=128 -> 2 strips)
    {
        int units = (NUM_SEL / 16) * (DOUT / 64);
        gemm_wmma_f16<<<cdiv(units, 8), 256, 0, stream>>>(
            hg, hg, HID, wtG, bg, (float*)d_out, NUM_SEL, HID, DOUT);
    }
}
